// Model_61950608278140
// MI455X (gfx1250) — matrix operand order repaired and checked by reading, not compiled, not run on hardware
//
#include <hip/hip_runtime.h>
#include <cstdint>

typedef __attribute__((ext_vector_type(16))) __bf16       v16bf;
typedef __attribute__((ext_vector_type(8)))  float        v8f;
typedef __attribute__((ext_vector_type(8)))  unsigned int v8u;
typedef __attribute__((ext_vector_type(4)))  unsigned int u32x4;
typedef __attribute__((ext_vector_type(8)))  int          i32x8;
typedef __attribute__((ext_vector_type(4)))  int          i32x4;

#define BATCH 65536
#define KIN   500
#define KPAD  512
#define NOUT  50
#define NPAD  64
#define SSTR  528                 // LDS row stride in bf16 elems (1056 B): 32B-aligned,
                                  // +8-bank shift per row -> minimal 4-way bank traffic
#define ROWS_PER_WAVE   32
#define WAVES_PER_BLOCK 8
#define ROWS_PER_BLOCK  (ROWS_PER_WAVE * WAVES_PER_BLOCK)   // 256

#if __has_builtin(__builtin_amdgcn_tensor_load_to_lds) && \
    __has_builtin(__builtin_amdgcn_s_wait_tensorcnt)
#define USE_TDM 1
#else
#define USE_TDM 0
#endif

__device__ __forceinline__ unsigned short f32_to_bf16_rne(float f) {
    unsigned int u = __builtin_bit_cast(unsigned int, f);
    u += 0x7FFFu + ((u >> 16) & 1u);
    return (unsigned short)(u >> 16);
}

// Two f32 -> packed bf16x2 in 3 VALU: bias both words (round-half-up; the
// carry legally ripples into the exponent), then one v_perm_b32 grabs the
// two high halves: result = {hi16(hi+0x8000), hi16(lo+0x8000)}.
__device__ __forceinline__ unsigned int pack_bf16(float lo, float hi) {
    unsigned int ulo = __builtin_bit_cast(unsigned int, lo) + 0x8000u;
    unsigned int uhi = __builtin_bit_cast(unsigned int, hi) + 0x8000u;
    return __builtin_amdgcn_perm(uhi, ulo, 0x07060302u);
}

// ---------------------------------------------------------------------------
// Kernel 1: contract MPO cores into W1 (500x50 -> padded, transposed, bf16),
// padded b1[64], padded W2[64][2]. Runs once per launch; ~8 MFLOP total.
// W1T layout: [n][k], n in [0,64), k in [0,512), bf16.
// ---------------------------------------------------------------------------
__global__ __launch_bounds__(256) void tn_precompute(
    const float* __restrict__ A1, const float* __restrict__ A2,
    const float* __restrict__ A3, const float* __restrict__ A4,
    const float* __restrict__ A5, const float* __restrict__ b1,
    const float* __restrict__ B1, const float* __restrict__ B2,
    const float* __restrict__ B3,
    unsigned short* __restrict__ W1T, float* __restrict__ b1pad,
    float* __restrict__ W2pad)
{
    const int idx = blockIdx.x * 256 + threadIdx.x;

    if (idx < NPAD * KPAD) {
        const int o = idx >> 9;    // output feature (n), row of W1T
        const int f = idx & 511;   // input feature (k)
        float wval = 0.f;
        if (o < NOUT && f < KIN) {
            // f = (((i*5+j)*5+k)*5+l)*2+m ; o = (v*5+w)*2+x
            int m = f % 2, t = f / 2;
            int l = t % 5; t /= 5;
            int k = t % 5; t /= 5;
            int j = t % 5; int i = t / 5;
            int x = o % 2; int t2 = o / 2;
            int w = t2 % 5; int v = t2 / 5;

            float a2[10], a3[10], a4[10];
            #pragma unroll
            for (int q = 0; q < 10; ++q) {
                float s = 0.f;
                #pragma unroll
                for (int p = 0; p < 10; ++p)
                    s += A1[i * 10 + p] * A2[p * 250 + j * 50 + q * 5 + v];
                a2[q] = s;
            }
            #pragma unroll
            for (int r = 0; r < 10; ++r) {
                float s = 0.f;
                #pragma unroll
                for (int q = 0; q < 10; ++q)
                    s += a2[q] * A3[q * 250 + k * 50 + r * 5 + w];
                a3[r] = s;
            }
            #pragma unroll
            for (int s2 = 0; s2 < 10; ++s2) {
                float s = 0.f;
                #pragma unroll
                for (int r = 0; r < 10; ++r)
                    s += a3[r] * A4[r * 100 + l * 20 + s2 * 2 + x];
                a4[s2] = s;
            }
            #pragma unroll
            for (int s2 = 0; s2 < 10; ++s2)
                wval += a4[s2] * A5[s2 * 2 + m];
        }
        W1T[idx] = f32_to_bf16_rne(wval);  // idx == o*KPAD + f
    }

    const int e = idx - NPAD * KPAD;
    if (e >= 0 && e < NPAD) {
        b1pad[e] = (e < NOUT) ? b1[e] : 0.f;
    } else if (e >= NPAD && e < NPAD + 2 * NPAD) {
        const int ee = e - NPAD;
        const int g = ee >> 1, v = ee & 1;
        float wval = 0.f;
        if (g < NOUT) {
            // g = (i*5+j)*5+k
            int k = g % 5, t = g / 5;
            int j = t % 5, i = t / 5;
            #pragma unroll
            for (int q = 0; q < 10; ++q) {
                float s = 0.f;
                #pragma unroll
                for (int p = 0; p < 10; ++p)
                    s += B1[i * 10 + p] * B2[p * 100 + j * 20 + q * 2 + v];
                wval += s * B3[q * 5 + k];
            }
        }
        W2pad[ee] = wval;
    }
}

// ---------------------------------------------------------------------------
// Load one 16x32 A tile (fp32 x -> packed bf16, contiguous K order, same as B).
// lane<16: K run {kb..kb+15} with kb=k0 ; lane>=16: kb=k0+16.
// Fast path: both runs fully in-bounds (kt <= 14).
// ---------------------------------------------------------------------------
__device__ __forceinline__ v16bf load_a_fast(const float* __restrict__ xrow,
                                             int kb) {
    const float4 a0 = *(const float4*)(xrow + kb);
    const float4 a1 = *(const float4*)(xrow + kb + 4);
    const float4 c0 = *(const float4*)(xrow + kb + 8);
    const float4 c1 = *(const float4*)(xrow + kb + 12);
    v8u ua;
    ua[0] = pack_bf16(a0.x, a0.y); ua[1] = pack_bf16(a0.z, a0.w);
    ua[2] = pack_bf16(a1.x, a1.y); ua[3] = pack_bf16(a1.z, a1.w);
    ua[4] = pack_bf16(c0.x, c0.y); ua[5] = pack_bf16(c0.z, c0.w);
    ua[6] = pack_bf16(c1.x, c1.y); ua[7] = pack_bf16(c1.z, c1.w);
    return __builtin_bit_cast(v16bf, ua);
}

// Tail (kt == 15): first 4 elements end at index 499 (always valid); the
// remaining 12 (max 511) need per-element masking against KIN.
__device__ __forceinline__ v16bf load_a_tail(const float* __restrict__ xrow,
                                             int kb) {
    const float4 a0 = *(const float4*)(xrow + kb);
    float fb[12];
    #pragma unroll
    for (int e2 = 0; e2 < 12; ++e2) {
        const int kk = kb + 4 + e2;
        fb[e2] = (kk < KIN) ? xrow[kk] : 0.f;   // W1T is 0 there anyway
    }
    v8u ua;
    ua[0] = pack_bf16(a0.x, a0.y); ua[1] = pack_bf16(a0.z, a0.w);
    ua[2] = pack_bf16(fb[0], fb[1]); ua[3] = pack_bf16(fb[2], fb[3]);
    ua[4] = pack_bf16(fb[4], fb[5]); ua[5] = pack_bf16(fb[6], fb[7]);
    ua[6] = pack_bf16(fb[8], fb[9]); ua[7] = pack_bf16(fb[10], fb[11]);
    return __builtin_bit_cast(v16bf, ua);
}

#define WMMA_STEP(A0, A1)                                                     \
    do {                                                                      \
        acc[0][0] = __builtin_amdgcn_wmma_f32_16x16x32_bf16(                  \
            false, (A0), false, bm0, (short)0, acc[0][0], false, false);      \
        acc[1][0] = __builtin_amdgcn_wmma_f32_16x16x32_bf16(                  \
            false, (A1), false, bm0, (short)0, acc[1][0], false, false);      \
        acc[0][1] = __builtin_amdgcn_wmma_f32_16x16x32_bf16(                  \
            false, (A0), false, bm1, (short)0, acc[0][1], false, false);      \
        acc[1][1] = __builtin_amdgcn_wmma_f32_16x16x32_bf16(                  \
            false, (A1), false, bm1, (short)0, acc[1][1], false, false);      \
        acc[0][2] = __builtin_amdgcn_wmma_f32_16x16x32_bf16(                  \
            false, (A0), false, bm2, (short)0, acc[0][2], false, false);      \
        acc[1][2] = __builtin_amdgcn_wmma_f32_16x16x32_bf16(                  \
            false, (A1), false, bm2, (short)0, acc[1][2], false, false);      \
        acc[0][3] = __builtin_amdgcn_wmma_f32_16x16x32_bf16(                  \
            false, (A0), false, bm3, (short)0, acc[0][3], false, false);      \
        acc[1][3] = __builtin_amdgcn_wmma_f32_16x16x32_bf16(                  \
            false, (A1), false, bm3, (short)0, acc[1][3], false, false);      \
    } while (0)

// ---------------------------------------------------------------------------
// Kernel 2: out = relu(x @ W1 + b1) @ W2 + b2.
// One wave = 32 rows x 64 (padded) cols; W1T staged in LDS by the Tensor Data
// Mover (row padded 1024B -> 1056B for bank spread); 15 branch-free K-steps
// of 32 with double-buffered B tiles + one peeled masked tail step.
// ---------------------------------------------------------------------------
__global__ __launch_bounds__(256) void tn_forward(
    const float* __restrict__ x,
    const unsigned short* __restrict__ W1T,
    const float* __restrict__ b1pad,
    const float* __restrict__ W2pad,
    const float* __restrict__ b2,
    float* __restrict__ out)
{
    // 64 rows x 528 elems (row stride 1056 B) = 66 KB of the 320 KB/WGP LDS
    __shared__ __align__(128) unsigned short sW[NPAD * SSTR];

#if USE_TDM
    if (threadIdx.x < 32) {   // wave 0 drives the TDM
        const unsigned long long ga = (unsigned long long)(uintptr_t)W1T;
        const unsigned int      lds = (unsigned int)(uintptr_t)(&sW[0]);
        u32x4 g0;
        g0[0] = 1u;                                         // count=1, user D#
        g0[1] = lds;                                        // lds_addr
        g0[2] = (unsigned int)(ga & 0xffffffffu);           // global_addr[31:0]
        g0[3] = (unsigned int)((ga >> 32) & 0x01ffffffu)    // global_addr[56:32]
              | (2u << 30);                                 // type = 2 (image)
        i32x8 g1;
        g1[0] = (int)((3u << 16)      // data_size = 8 bytes
                    | (1u << 20)      // pad_enable
                    | (7u << 22)      // pad_interval: 256 dwords (one 1024B row)
                    | (7u << 25));    // pad_amount:   8 dwords (32 B)
        g1[1] = (int)(128u << 16);    // tensor_dim0 = 128 (8B units) [bits 79:48]
        g1[2] = (int)(64u << 16);     // tensor_dim1 = 64            [bits 111:80]
        g1[3] = (int)(128u << 16);    // tile_dim0 = 128             [bits 127:112]
        g1[4] = (int)(64u);           // tile_dim1 = 64, tile_dim2 = 0
        g1[5] = (int)128;             // tensor_dim0_stride = 128 (8B units)
        g1[6] = 0;                    // dim0_stride hi, dim1_stride lo
        g1[7] = 0;                    // dim1_stride hi (unused, 2D tile)
        const i32x4 z4 = {};
        const i32x8 z8 = {};
        // 6-arg form (clang-23 / therock-10.0): groups 2 & 3 unused (2D tile)
        __builtin_amdgcn_tensor_load_to_lds(g0, g1, z4, z4, z8, 0);
        __builtin_amdgcn_s_wait_tensorcnt(0);
    }
#else
    {   // fallback: cooperative fill, same padded layout
        const int tid  = threadIdx.x;
        const int row  = tid >> 2;
        const int part = tid & 3;
        #pragma unroll
        for (int c = 0; c < 16; ++c) {
            const int off = part * 128 + c * 8;          // element offset in row
            const uint4 v = *(const uint4*)(W1T + row * KPAD + off);
            *(uint4*)(&sW[row * SSTR + off]) = v;
        }
    }
#endif
    __syncthreads();

    const int lane  = threadIdx.x & 31;
    const int wave  = threadIdx.x >> 5;
    const int lrow  = lane & 15;   // A: M row ; B/C: N column (mod 16)
    const int lhalf = lane >> 4;   // K half-select for A/B ; M half for C
    const int m0    = blockIdx.x * ROWS_PER_BLOCK + wave * ROWS_PER_WAVE;

    const float* xrow0 = x + (size_t)(m0 + lrow) * KIN;
    const float* xrow1 = xrow0 + (size_t)16 * KIN;

    // B-tile LDS element offsets: n = t*16 + lrow, K run starts at lhalf*16
    int nb0, nb1, nb2, nb3;
    {
        const int koff = lhalf * 16;
        nb0 = (0 * 16 + lrow) * SSTR + koff;
        nb1 = (1 * 16 + lrow) * SSTR + koff;
        nb2 = (2 * 16 + lrow) * SSTR + koff;
        nb3 = (3 * 16 + lrow) * SSTR + koff;
    }

    v8f acc[2][4];
    #pragma unroll
    for (int g = 0; g < 2; ++g)
        #pragma unroll
        for (int t = 0; t < 4; ++t)
            acc[g][t] = (v8f){};

    // prime B double-buffer with kt = 0
    v16bf bm0 = *(const v16bf*)(&sW[nb0]);
    v16bf bm1 = *(const v16bf*)(&sW[nb1]);
    v16bf bm2 = *(const v16bf*)(&sW[nb2]);
    v16bf bm3 = *(const v16bf*)(&sW[nb3]);

    // ---- 15 branch-free K-steps (all loads provably in-bounds)
    #pragma unroll 3
    for (int kt = 0; kt < 15; ++kt) {
        const int kb = kt * 32 + lhalf * 16;

        __builtin_prefetch(xrow0 + kb + 64, 0, 3);       // global_prefetch_b8
        __builtin_prefetch(xrow1 + kb + 64, 0, 3);

        const v16bf a0 = load_a_fast(xrow0, kb);
        const v16bf a1 = load_a_fast(xrow1, kb);

        // issue next K-step's B tiles now; latency hides under the 8 WMMAs
        const int nk = (kt + 1) * 32;
        const v16bf bn0 = *(const v16bf*)(&sW[nb0 + nk]);
        const v16bf bn1 = *(const v16bf*)(&sW[nb1 + nk]);
        const v16bf bn2 = *(const v16bf*)(&sW[nb2 + nk]);
        const v16bf bn3 = *(const v16bf*)(&sW[nb3 + nk]);

        WMMA_STEP(a0, a1);

        bm0 = bn0; bm1 = bn1; bm2 = bn2; bm3 = bn3;
    }

    // ---- peeled tail K-step (kt = 15): masked second A-run, B already live
    {
        const int kb = 15 * 32 + lhalf * 16;
        const v16bf a0 = load_a_tail(xrow0, kb);
        const v16bf a1 = load_a_tail(xrow1, kb);
        WMMA_STEP(a0, a1);
    }

    // -------- Epilogue: bias + relu + (50 -> 2) GEMM, fused in-register.
    // C layout: lane holds column n = t*16+lrow, rows M = lhalf*8 + r.
    float bias[4], w20[4], w21[4];
    #pragma unroll
    for (int t = 0; t < 4; ++t) {
        const int n = t * 16 + lrow;
        bias[t] = b1pad[n];
        w20[t]  = W2pad[n * 2 + 0];
        w21[t]  = W2pad[n * 2 + 1];
    }
    const float c0 = b2[0], c1 = b2[1];

    #pragma unroll
    for (int g = 0; g < 2; ++g) {
        #pragma unroll
        for (int r = 0; r < 8; ++r) {
            float s0 = 0.f, s1 = 0.f;
            #pragma unroll
            for (int t = 0; t < 4; ++t) {
                const float h = fmaxf(acc[g][t][r] + bias[t], 0.f);
                s0 += h * w20[t];
                s1 += h * w21[t];
            }
            // reduce across the 16 lanes sharing this row (xor<16 stays in half)
            #pragma unroll
            for (int msk = 1; msk <= 8; msk <<= 1) {
                s0 += __shfl_xor(s0, msk, 32);
                s1 += __shfl_xor(s1, msk, 32);
            }
            if (lrow == 0) {
                const int row = m0 + g * 16 + lhalf * 8 + r;
                *(float2*)(out + (size_t)row * 2) = make_float2(s0 + c0, s1 + c1);
            }
        }
    }
}

// ---------------------------------------------------------------------------
extern "C" void kernel_launch(void* const* d_in, const int* in_sizes, int n_in,
                              void* d_out, int out_size, void* d_ws, size_t ws_size,
                              hipStream_t stream) {
    (void)in_sizes; (void)n_in; (void)out_size; (void)ws_size;
    const float* x  = (const float*)d_in[0];
    const float* A1 = (const float*)d_in[1];
    const float* A2 = (const float*)d_in[2];
    const float* A3 = (const float*)d_in[3];
    const float* A4 = (const float*)d_in[4];
    const float* A5 = (const float*)d_in[5];
    const float* b1 = (const float*)d_in[6];
    const float* B1 = (const float*)d_in[7];
    const float* B2 = (const float*)d_in[8];
    const float* B3 = (const float*)d_in[9];
    const float* b2 = (const float*)d_in[10];

    unsigned short* W1T  = (unsigned short*)d_ws;                 // 64*512*2 = 65536 B
    float*          b1pd = (float*)((char*)d_ws + 65536);         // 64 floats
    float*          W2pd = b1pd + NPAD;                           // 64*2 floats

    tn_precompute<<<129, 256, 0, stream>>>(A1, A2, A3, A4, A5, b1,
                                           B1, B2, B3, W1T, b1pd, W2pd);

    tn_forward<<<BATCH / ROWS_PER_BLOCK, 256, 0, stream>>>(
        x, W1T, b1pd, W2pd, b2, (float*)d_out);
}
